// MultiHeadAttention_7550552507184
// MI455X (gfx1250) — matrix operand order repaired and checked by reading, not compiled, not run on hardware
//
#include <hip/hip_runtime.h>
#include <hip/hip_bf16.h>

typedef __attribute__((ext_vector_type(8)))  float  v8f;
typedef __attribute__((ext_vector_type(16))) __bf16 v16bf;
typedef __attribute__((ext_vector_type(8)))  __bf16 v8bf;

#define BB  2
#define SS  2048
#define DD  1024
#define HH  16
#define HDD 64

static __device__ __forceinline__ __bf16 f2bf(float f) {
  unsigned u = __builtin_bit_cast(unsigned, f);
  unsigned r = (u + 0x7FFFu + ((u >> 16) & 1u)) >> 16;   // round-to-nearest-even
  return __builtin_bit_cast(__bf16, (unsigned short)r);
}

static __device__ __forceinline__ v16bf cat16(v8bf lo, v8bf hi) {
  v16bf r;
#pragma unroll
  for (int i = 0; i < 8; ++i) { r[i] = lo[i]; r[i + 8] = hi[i]; }
  return r;
}

static __device__ __forceinline__ v8f zero8() {
  v8f z;
#pragma unroll
  for (int i = 0; i < 8; ++i) z[i] = 0.f;
  return z;
}

#define WMMA_BF16(a, b, c) \
  __builtin_amdgcn_wmma_f32_16x16x32_bf16(false, (a), false, (b), (short)0, (c), false, false)

// --------- CDNA5 async copy (global -> LDS, 16B per lane, tracked by ASYNCcnt)
// Inline asm per CDNA5 ISA: VDST = LDS byte address (AS3),
// VADDR = 64-bit global address, SADDR = off (GV mode).
typedef __attribute__((address_space(3))) char* lds_ptr_t;

static __device__ __forceinline__ void async_cp_b128(const void* g, void* l) {
  lds_ptr_t l3 = (lds_ptr_t)l;
  asm volatile("global_load_async_to_lds_b128 %0, %1, off"
               : : "v"(l3), "v"(g) : "memory");
}

static __device__ __forceinline__ void wait_async0() {
#if __has_builtin(__builtin_amdgcn_s_wait_asynccnt)
  __builtin_amdgcn_s_wait_asynccnt(0);
#else
  asm volatile("s_wait_asynccnt 0" ::: "memory");
#endif
}

// ---------------------------------------------------------------- prep kernels
__global__ void k_cvt(const float* __restrict__ x, __bf16* __restrict__ out, int n) {
  int i = blockIdx.x * blockDim.x + threadIdx.x;
  if (i < n) out[i] = f2bf(x[i]);
}

// W[k][n] f32 -> WT[n][k] bf16 (so WMMA B-fragments load K-contiguous)
__global__ void k_transpose(const float* __restrict__ w, __bf16* __restrict__ wt) {
  int i = blockIdx.x * blockDim.x + threadIdx.x;   // i = n*1024 + k
  int n = i >> 10, k = i & 1023;
  wt[i] = f2bf(w[k * 1024 + n]);
}

// ---------------------------------------------------- fused QKV projection GEMM
// Block = 8 waves x (16 rows each) sharing one 64-col weight panel for Q,K,V.
// Weight panels are double-buffered in LDS via async global->LDS copies.
__global__ void __launch_bounds__(256) k_qkv(const __bf16* __restrict__ xb,
                                             const __bf16* __restrict__ wqT,
                                             const __bf16* __restrict__ wkT,
                                             const __bf16* __restrict__ wvT,
                                             __bf16* __restrict__ Q,
                                             __bf16* __restrict__ K,
                                             __bf16* __restrict__ VT) {
  __shared__ __align__(16) __bf16 bufB[2][3 * 2048];   // 2 x 12KB panels
  const int tid   = threadIdx.x;
  const int wslot = tid >> 5;
  const int lane  = tid & 31, half = lane >> 4, lm = lane & 15;
  const int cg    = blockIdx.x & 15;        // 16 col groups of 64
  const int rblk  = blockIdx.x >> 4;        // 32 row blocks of 8x16 rows
  const int r0    = (rblk * 8 + wslot) * 16;
  const int n0    = cg * 64;
  const __bf16* wT[3] = { wqT, wkT, wvT };

  const int cn = tid >> 2, cs = tid & 3;    // copy lane mapping: col, 16B segment

  // prologue: stage K-chunk 0
#pragma unroll
  for (int w = 0; w < 3; ++w)
    async_cp_b128(wT[w] + (size_t)(n0 + cn) * DD + cs * 8,
                  &bufB[0][w * 2048 + cn * 32 + cs * 8]);
  wait_async0();
  __syncthreads();

  v8f acc[3][4];
#pragma unroll
  for (int w = 0; w < 3; ++w)
#pragma unroll
    for (int t = 0; t < 4; ++t) acc[w][t] = zero8();

  const __bf16* arow = xb + (size_t)(r0 + lm) * DD;

  for (int kc = 0; kc < 32; ++kc) {
    const int cur = kc & 1, nxt = cur ^ 1;
    if (kc + 1 < 32) {
      const int k2 = (kc + 1) * 32;
#pragma unroll
      for (int w = 0; w < 3; ++w)
        async_cp_b128(wT[w] + (size_t)(n0 + cn) * DD + k2 + cs * 8,
                      &bufB[nxt][w * 2048 + cn * 32 + cs * 8]);
    }
    const int k = kc * 32;
    v16bf a = cat16(*(const v8bf*)(arow + k + 8 * half),
                    *(const v8bf*)(arow + k + 16 + 8 * half));
    const __bf16* base = &bufB[cur][0];
#pragma unroll
    for (int w = 0; w < 3; ++w) {
#pragma unroll
      for (int t = 0; t < 4; ++t) {
        const __bf16* bp = base + w * 2048 + (t * 16 + lm) * 32 + 8 * half;
        v16bf b = cat16(*(const v8bf*)bp, *(const v8bf*)(bp + 16));
        acc[w][t] = WMMA_BF16(a, b, acc[w][t]);
      }
    }
    wait_async0();     // our share of next panel landed
    __syncthreads();   // everyone done reading cur + all copies to nxt visible
  }

#pragma unroll
  for (int t = 0; t < 4; ++t) {
    const int col = n0 + t * 16 + lm;
    const int h = col >> 6, hd = col & 63;
#pragma unroll
    for (int i = 0; i < 8; ++i) {
      const int r = r0 + i + 8 * half;
      const int b = r >> 11, s = r & 2047;
      const size_t idx = (((size_t)(b * HH + h) * SS + s) * HDD + hd);
      Q[idx] = f2bf(acc[0][t][i]);
      K[idx] = f2bf(acc[1][t][i]);
    }
    // V stored transposed [b][h][hd][s]; lane holds fixed hd, 8 consecutive s -> packed store
    v8bf pv;
#pragma unroll
    for (int i = 0; i < 8; ++i) pv[i] = f2bf(acc[2][t][i]);
    const int rb = r0 + 8 * half;
    const int b = rb >> 11, s = rb & 2047;
    *(v8bf*)(VT + (((size_t)(b * HH + h) * HDD + hd) * SS + s)) = pv;
  }
}

// --------------------------------------------------------------- flash attention
// one wave per 16-row q tile; key blocks of 32; online softmax; P via LDS.
__global__ void __launch_bounds__(128) k_flash(const __bf16* __restrict__ Q,
                                               const __bf16* __restrict__ K,
                                               const __bf16* __restrict__ VT,
                                               __bf16* __restrict__ ctx) {
  __shared__ __align__(16) __bf16 Plds[4 * 16 * 32];
  const int wslot = threadIdx.x >> 5;
  const int wid   = blockIdx.x * 4 + wslot;
  const int lane  = threadIdx.x & 31, half = lane >> 4, lm = lane & 15;
  const int b   = wid >> 11;          // 2048 q-tiles per batch (16 heads * 128 tiles)
  const int rem = wid & 2047;
  const int h   = rem >> 7;
  const int q0  = (rem & 127) * 16;

  const size_t bh = (size_t)(b * HH + h);
  const __bf16* Qb = Q  + bh * SS * HDD;
  const __bf16* Kb = K  + bh * SS * HDD;
  const __bf16* Vb = VT + bh * HDD * SS;
  __bf16* Pw = Plds + wslot * (16 * 32);

  v16bf aq[2];
#pragma unroll
  for (int c = 0; c < 2; ++c) {
    const __bf16* p = Qb + (size_t)(q0 + lm) * HDD + c * 32 + 8 * half;
    aq[c] = cat16(*(const v8bf*)p, *(const v8bf*)(p + 16));
  }

  float mrow[8], lrow[8];
#pragma unroll
  for (int i = 0; i < 8; ++i) { mrow[i] = -3.0e38f; lrow[i] = 0.f; }
  v8f o[4];
#pragma unroll
  for (int c = 0; c < 4; ++c) o[c] = zero8();

  const int qEnd = q0 + 16;
  for (int s0 = 0; s0 < qEnd; s0 += 32) {
    // prefetch next key/value block (global_prefetch_b8)
    if (s0 + 32 < qEnd) {
      __builtin_prefetch(Kb + (size_t)(s0 + 32 + lane) * HDD, 0, 1);
      __builtin_prefetch(Vb + (size_t)(lane * 2) * SS + s0 + 32, 0, 1);
    }
    v8f acc0 = zero8(), acc1 = zero8();
#pragma unroll
    for (int c = 0; c < 2; ++c) {
      const __bf16* kp0 = Kb + (size_t)(s0 + lm) * HDD + c * 32 + 8 * half;
      v16bf bk0 = cat16(*(const v8bf*)kp0, *(const v8bf*)(kp0 + 16));
      acc0 = WMMA_BF16(aq[c], bk0, acc0);
      const __bf16* kp1 = Kb + (size_t)(s0 + 16 + lm) * HDD + c * 32 + 8 * half;
      v16bf bk1 = cat16(*(const v8bf*)kp1, *(const v8bf*)(kp1 + 16));
      acc1 = WMMA_BF16(aq[c], bk1, acc1);
    }

    const int col0 = s0 + lm, col1 = s0 + 16 + lm;
    float p0[8], p1[8], scl[8];
#pragma unroll
    for (int i = 0; i < 8; ++i) {
      const int rowg = q0 + i + 8 * half;
      float v0 = (col0 <= rowg) ? acc0[i] * 0.125f : -3.0e38f;  // 1/sqrt(64)
      float v1 = (col1 <= rowg) ? acc1[i] * 0.125f : -3.0e38f;
      float mb = fmaxf(v0, v1);
      for (int d = 1; d < 16; d <<= 1) mb = fmaxf(mb, __shfl_xor(mb, d, 32));
      const float mnew = fmaxf(mrow[i], mb);
      scl[i] = __expf(mrow[i] - mnew);
      p0[i]  = __expf(v0 - mnew);
      p1[i]  = __expf(v1 - mnew);
      float rs = p0[i] + p1[i];
      for (int d = 1; d < 16; d <<= 1) rs += __shfl_xor(rs, d, 32);
      lrow[i] = lrow[i] * scl[i] + rs;
      mrow[i] = mnew;
    }

    // stage P (16x32 bf16) to LDS row-major, read back in A-fragment layout
#pragma unroll
    for (int i = 0; i < 8; ++i) {
      const int r = i + 8 * half;
      Pw[r * 32 + lm]      = f2bf(p0[i]);
      Pw[r * 32 + 16 + lm] = f2bf(p1[i]);
    }
    asm volatile("s_wait_dscnt 0" ::: "memory");  // same-wave RAW through LDS
    const __bf16* pp = Pw + lm * 32 + 8 * half;
    v16bf pf = cat16(*(const v8bf*)pp, *(const v8bf*)(pp + 16));

#pragma unroll
    for (int c = 0; c < 4; ++c) {
#pragma unroll
      for (int i = 0; i < 8; ++i) o[c][i] *= scl[i];
      const __bf16* vp = Vb + (size_t)(c * 16 + lm) * SS + s0 + 8 * half;
      v16bf bv = cat16(*(const v8bf*)vp, *(const v8bf*)(vp + 16));
      o[c] = WMMA_BF16(pf, bv, o[c]);
    }
  }

  float inv[8];
#pragma unroll
  for (int i = 0; i < 8; ++i) inv[i] = 1.0f / lrow[i];
#pragma unroll
  for (int c = 0; c < 4; ++c) {
    const int col = h * HDD + c * 16 + lm;
#pragma unroll
    for (int i = 0; i < 8; ++i) {
      const int row = b * SS + q0 + i + 8 * half;
      ctx[(size_t)row * DD + col] = f2bf(o[c][i] * inv[i]);
    }
  }
}

// ---------------------------------------------------------- output projection
// Block = 8 waves sharing one async-staged, double-buffered WoT panel.
__global__ void __launch_bounds__(256) k_oproj(const __bf16* __restrict__ ctx,
                                               const __bf16* __restrict__ woT,
                                               const float* __restrict__ bo,
                                               float* __restrict__ out) {
  __shared__ __align__(16) __bf16 bufB[2][2048];   // 2 x 4KB panels
  const int tid   = threadIdx.x;
  const int wslot = tid >> 5;
  const int lane  = tid & 31, half = lane >> 4, lm = lane & 15;
  const int cg    = blockIdx.x & 15;
  const int rblk  = blockIdx.x >> 4;
  const int r0    = (rblk * 8 + wslot) * 16;
  const int n0    = cg * 64;
  const int cn = tid >> 2, cs = tid & 3;

  async_cp_b128(woT + (size_t)(n0 + cn) * DD + cs * 8, &bufB[0][cn * 32 + cs * 8]);
  wait_async0();
  __syncthreads();

  v8f acc[4];
#pragma unroll
  for (int t = 0; t < 4; ++t) acc[t] = zero8();

  const __bf16* arow = ctx + (size_t)(r0 + lm) * DD;
  for (int kc = 0; kc < 32; ++kc) {
    const int cur = kc & 1, nxt = cur ^ 1;
    if (kc + 1 < 32)
      async_cp_b128(woT + (size_t)(n0 + cn) * DD + (kc + 1) * 32 + cs * 8,
                    &bufB[nxt][cn * 32 + cs * 8]);
    const int k = kc * 32;
    v16bf a = cat16(*(const v8bf*)(arow + k + 8 * half),
                    *(const v8bf*)(arow + k + 16 + 8 * half));
#pragma unroll
    for (int t = 0; t < 4; ++t) {
      const __bf16* bp = &bufB[cur][(t * 16 + lm) * 32 + 8 * half];
      v16bf b = cat16(*(const v8bf*)bp, *(const v8bf*)(bp + 16));
      acc[t] = WMMA_BF16(a, b, acc[t]);
    }
    wait_async0();
    __syncthreads();
  }
#pragma unroll
  for (int t = 0; t < 4; ++t) {
    const int col = n0 + t * 16 + lm;
    const float bias = bo[col];
#pragma unroll
    for (int i = 0; i < 8; ++i) {
      const int r = r0 + i + 8 * half;
      out[(size_t)r * DD + col] = acc[t][i] + bias;
    }
  }
}

// ------------------------------------------------------------------- launcher
extern "C" void kernel_launch(void* const* d_in, const int* in_sizes, int n_in,
                              void* d_out, int out_size, void* d_ws, size_t ws_size,
                              hipStream_t stream) {
  (void)in_sizes; (void)n_in; (void)out_size; (void)ws_size;
  const float* x  = (const float*)d_in[0];
  const float* Wq = (const float*)d_in[1];
  const float* Wk = (const float*)d_in[2];
  const float* Wv = (const float*)d_in[3];
  const float* Wo = (const float*)d_in[4];
  const float* bo = (const float*)d_in[5];
  float* out = (float*)d_out;

  char* w = (char*)d_ws;
  __bf16* xbf = (__bf16*)w;  w += (size_t)4096 * 1024 * 2;
  __bf16* wqT = (__bf16*)w;  w += (size_t)1024 * 1024 * 2;
  __bf16* wkT = (__bf16*)w;  w += (size_t)1024 * 1024 * 2;
  __bf16* wvT = (__bf16*)w;  w += (size_t)1024 * 1024 * 2;
  __bf16* woT = (__bf16*)w;  w += (size_t)1024 * 1024 * 2;
  __bf16* Qb  = (__bf16*)w;  w += (size_t)BB * HH * SS * HDD * 2;
  __bf16* Kb  = (__bf16*)w;  w += (size_t)BB * HH * SS * HDD * 2;
  __bf16* VTb = (__bf16*)w;  w += (size_t)BB * HH * SS * HDD * 2;
  __bf16* ctx = (__bf16*)w;  w += (size_t)4096 * 1024 * 2;

  k_cvt<<<16384, 256, 0, stream>>>(x, xbf, 4096 * 1024);
  k_transpose<<<4096, 256, 0, stream>>>(Wq, wqT);
  k_transpose<<<4096, 256, 0, stream>>>(Wk, wkT);
  k_transpose<<<4096, 256, 0, stream>>>(Wv, wvT);
  k_transpose<<<4096, 256, 0, stream>>>(Wo, woT);
  k_qkv<<<512, 256, 0, stream>>>(xbf, wqT, wkT, wvT, Qb, Kb, VTb);
  k_flash<<<1024, 128, 0, stream>>>(Qb, Kb, VTb, ctx);
  k_oproj<<<512, 256, 0, stream>>>(ctx, woT, bo, out);
}
